// MultiHeadAttentionClassical_65481071407421
// MI455X (gfx1250) — matrix operand order repaired and checked by reading, not compiled, not run on hardware
//
#include <hip/hip_runtime.h>

#define BATCH 2
#define SEQ   2048
#define EMBED 1024
#define HEADS 16
#define HDIM  64

typedef __attribute__((ext_vector_type(16))) __bf16 v16bf;
typedef __attribute__((ext_vector_type(8)))  __bf16 v8bf;
typedef __attribute__((ext_vector_type(8)))  float  v8f;

static __device__ __forceinline__ __bf16 to_bf(float f) { return (__bf16)f; }

// A-operand (16x32, MxK): lane L holds row L%16; group g=L/16 holds
// K in {g*8..g*8+7} (elements 0..7) and {16+g*8..} (elements 8..15).
static __device__ __forceinline__ v16bf load_a_tile(const __bf16* __restrict__ base, int ld) {
  const int lane = threadIdx.x & 31;
  const int row  = lane & 15;
  const int g    = lane >> 4;
  const __bf16* p = base + (size_t)row * ld + g * 8;
  v8bf c0 = *(const v8bf*)(p);
  v8bf c1 = *(const v8bf*)(p + 16);
  v16bf r;
#pragma unroll
  for (int i = 0; i < 8; ++i) { r[i] = c0[i]; r[i + 8] = c1[i]; }
  return r;
}

// B-operand (32x16, KxN) with memory holding N rows of contiguous K
// (i.e. computing X @ W^T): lane L holds col L%16, K in {g*8..g*8+7} and {16+g*8..}.
static __device__ __forceinline__ v16bf load_b_tile(const __bf16* __restrict__ base, int ld) {
  const int lane = threadIdx.x & 31;
  const int col  = lane & 15;
  const int g    = lane >> 4;
  const __bf16* p = base + (size_t)col * ld + g * 8;
  v8bf c0 = *(const v8bf*)(p);
  v8bf c1 = *(const v8bf*)(p + 16);
  v16bf r;
#pragma unroll
  for (int i = 0; i < 8; ++i) { r[i] = c0[i]; r[i + 8] = c1[i]; }
  return r;
}

static __device__ __forceinline__ v8f wmma_bf16(v16bf a, v16bf b, v8f c) {
  return __builtin_amdgcn_wmma_f32_16x16x32_bf16(false, a, false, b, (short)0, c, false, false);
}

static __device__ __forceinline__ void zero8(v8f& v) {
#pragma unroll
  for (int i = 0; i < 8; ++i) v[i] = 0.0f;
}

// ---------------- fp32 -> bf16 conversion ----------------
__global__ void cvt_f32_bf16(const float* __restrict__ src, __bf16* __restrict__ dst, int n) {
  int i = blockIdx.x * blockDim.x + threadIdx.x;
  if (i < n) dst[i] = to_bf(src[i]);
}

// ---------------- QKV projection: (B*S,1024) @ (3072,1024)^T ----------------
// Each wave computes a 16x64 output tile, software-pipelined over K.
// Q scaled by 1/sqrt(HDIM)=0.125.  Q,K stored (b,h,s,d); V transposed (b,h,d,s).
__global__ __launch_bounds__(256) void qkv_gemm(const __bf16* __restrict__ xh,
                                                const __bf16* __restrict__ wh,
                                                __bf16* __restrict__ Qh,
                                                __bf16* __restrict__ Kh,
                                                __bf16* __restrict__ Vt) {
  const int wave = blockIdx.x * 8 + (threadIdx.x >> 5);
  const int mt = wave / 48;          // 4096/16 = 256 row tiles
  const int ns = wave % 48;          // 3072/64 = 48 col strips
  const int lane = threadIdx.x & 31;
  const int g = lane >> 4, c = lane & 15;

  const __bf16* abase = xh + (size_t)mt * 16 * EMBED;
  const __bf16* bbase[4];
#pragma unroll
  for (int t = 0; t < 4; ++t) bbase[t] = wh + (size_t)(ns * 64 + t * 16) * EMBED;

  v8f acc[4];
#pragma unroll
  for (int t = 0; t < 4; ++t) zero8(acc[t]);

  // prologue loads (k = 0)
  v16bf a_cur = load_a_tile(abase, EMBED);
  v16bf b_cur[4];
#pragma unroll
  for (int t = 0; t < 4; ++t) b_cur[t] = load_b_tile(bbase[t], EMBED);

  // steady state: issue loads for k, consume tiles of k-32.
  // unroll 2 so the pipeline rotation becomes SSA renames, not v_mov copies.
#pragma unroll 2
  for (int k = 32; k < EMBED; k += 32) {
    v16bf a_nxt = load_a_tile(abase + k, EMBED);
    v16bf b_nxt[4];
#pragma unroll
    for (int t = 0; t < 4; ++t) b_nxt[t] = load_b_tile(bbase[t] + k, EMBED);
#pragma unroll
    for (int t = 0; t < 4; ++t) acc[t] = wmma_bf16(a_cur, b_cur[t], acc[t]);
    a_cur = a_nxt;
#pragma unroll
    for (int t = 0; t < 4; ++t) b_cur[t] = b_nxt[t];
  }
#pragma unroll
  for (int t = 0; t < 4; ++t) acc[t] = wmma_bf16(a_cur, b_cur[t], acc[t]);

#pragma unroll
  for (int t = 0; t < 4; ++t) {
    const int n     = ns * 64 + t * 16 + c;
    const int which = n >> 10;             // 0=Q 1=K 2=V
    const int hh    = (n >> 6) & (HEADS - 1);
    const int d     = n & 63;
#pragma unroll
    for (int r = 0; r < 8; ++r) {
      const int m = mt * 16 + r + 8 * g;
      const int b = m >> 11;               // / SEQ
      const int s = m & (SEQ - 1);
      const float v = acc[t][r];
      if (which == 0)
        Qh[(((size_t)(b * HEADS + hh) * SEQ + s) << 6) + d] = to_bf(v * 0.125f);
      else if (which == 1)
        Kh[(((size_t)(b * HEADS + hh) * SEQ + s) << 6) + d] = to_bf(v);
      else
        Vt[((size_t)(b * HEADS + hh) * HDIM + d) * SEQ + s] = to_bf(v);
    }
  }
}

// ---------------- fused attention: softmax(Q K^T) V ----------------
// grid(x)=B*HEADS, grid(y)=SEQ/64. 4 waves/block, 16 rows per wave.
__global__ __launch_bounds__(128) void attn_kernel(const __bf16* __restrict__ Qh,
                                                   const __bf16* __restrict__ Kh,
                                                   const __bf16* __restrict__ Vt,
                                                   const int* __restrict__ mask,
                                                   __bf16* __restrict__ Oh) {
  __shared__ __bf16 ldsP[4 * 16 * 32];          // 1KB per wave, wave-private
  const int bh = blockIdx.x;
  const int b  = bh >> 4;
  const int h  = bh & 15;
  const int w  = threadIdx.x >> 5;
  const int lane = threadIdx.x & 31;
  const int g = lane >> 4, c = lane & 15;
  const int row0 = blockIdx.y * 64 + w * 16;

  const __bf16* qbase = Qh + ((size_t)bh * SEQ + row0) * HDIM;
  const __bf16* kbase = Kh + (size_t)bh * SEQ * HDIM;
  const __bf16* vbase = Vt + (size_t)bh * HDIM * SEQ;
  const int* mp = mask + b * SEQ;

  const v16bf qa0 = load_a_tile(qbase, HDIM);        // K = 0..31
  const v16bf qa1 = load_a_tile(qbase + 32, HDIM);   // K = 32..63

  const float NEGINF = -__builtin_inff();
  float rmax[8];
#pragma unroll
  for (int r = 0; r < 8; ++r) rmax[r] = NEGINF;

  // ---- Pass 1: row maxima of masked scores (pipelined over K-row tiles)
  v16bf kb0 = load_b_tile(kbase, HDIM);
  v16bf kb1 = load_b_tile(kbase + 32, HDIM);
#pragma unroll 2
  for (int j = 0; j < SEQ; j += 16) {
    v16bf kn0, kn1;
    if (j + 16 < SEQ) {
      kn0 = load_b_tile(kbase + (size_t)(j + 16) * HDIM, HDIM);
      kn1 = load_b_tile(kbase + (size_t)(j + 16) * HDIM + 32, HDIM);
    }
    v8f s; zero8(s);
    s = wmma_bf16(qa0, kb0, s);
    s = wmma_bf16(qa1, kb1, s);
    const bool mv = (mp[j + c] != 0);
#pragma unroll
    for (int r = 0; r < 8; ++r) {
      float sv = mv ? s[r] : NEGINF;
      rmax[r] = fmaxf(rmax[r], sv);
    }
    kb0 = kn0; kb1 = kn1;
  }
#pragma unroll
  for (int r = 0; r < 8; ++r) {
    float v = rmax[r];
#pragma unroll
    for (int off = 8; off > 0; off >>= 1) v = fmaxf(v, __shfl_xor(v, off, 32));
    rmax[r] = v;
  }

  // ---- Pass 2: recompute scores, exp, accumulate P @ V
  float rsum[8];
#pragma unroll
  for (int r = 0; r < 8; ++r) rsum[r] = 0.0f;
  v8f o[4];
#pragma unroll
  for (int t = 0; t < 4; ++t) zero8(o[t]);

  __bf16* pbuf = ldsP + w * 512;

  for (int j = 0; j < SEQ; j += 32) {
    // issue ALL global loads for this iteration up front: K tiles for the
    // score WMMAs and V tiles that are only consumed after the LDS bounce.
    v16bf kb00 = load_b_tile(kbase + (size_t)j * HDIM, HDIM);
    v16bf kb01 = load_b_tile(kbase + (size_t)j * HDIM + 32, HDIM);
    v16bf kb10 = load_b_tile(kbase + (size_t)(j + 16) * HDIM, HDIM);
    v16bf kb11 = load_b_tile(kbase + (size_t)(j + 16) * HDIM + 32, HDIM);
    v16bf vb[4];
#pragma unroll
    for (int t = 0; t < 4; ++t) vb[t] = load_b_tile(vbase + (size_t)(t * 16) * SEQ + j, SEQ);
    const bool mv0 = (mp[j + c] != 0);
    const bool mv1 = (mp[j + 16 + c] != 0);

    v8f s0, s1; zero8(s0); zero8(s1);
    s0 = wmma_bf16(qa0, kb00, s0);
    s0 = wmma_bf16(qa1, kb01, s0);
    s1 = wmma_bf16(qa0, kb10, s1);
    s1 = wmma_bf16(qa1, kb11, s1);

#pragma unroll
    for (int r = 0; r < 8; ++r) {
      float p0 = mv0 ? __expf(s0[r] - rmax[r]) : 0.0f;
      float p1 = mv1 ? __expf(s1[r] - rmax[r]) : 0.0f;
      rsum[r] += p0 + p1;
      const int rr = r + 8 * g;
      pbuf[rr * 32 + c]      = to_bf(p0);   // P tile row-major 16x32 in LDS
      pbuf[rr * 32 + 16 + c] = to_bf(p1);
    }
    // same-wave DS ops are in-order: reload P as an A-operand tile
    v16bf pa = load_a_tile(pbuf, 32);
#pragma unroll
    for (int t = 0; t < 4; ++t) o[t] = wmma_bf16(pa, vb[t], o[t]);
  }

#pragma unroll
  for (int r = 0; r < 8; ++r) {
    float v = rsum[r];
#pragma unroll
    for (int off = 8; off > 0; off >>= 1) v += __shfl_xor(v, off, 32);
    rsum[r] = 1.0f / v;
  }

#pragma unroll
  for (int t = 0; t < 4; ++t) {
    const int e = h * 64 + t * 16 + c;
#pragma unroll
    for (int r = 0; r < 8; ++r) {
      const int s = row0 + r + 8 * g;
      Oh[((size_t)b * SEQ + s) * EMBED + e] = to_bf(o[t][r] * rsum[r]);
    }
  }
}

// ---------------- output projection: Oh @ out_w^T + out_b (fp32 out) ----------------
__global__ __launch_bounds__(256) void out_gemm(const __bf16* __restrict__ Oh,
                                                const __bf16* __restrict__ wh,
                                                const float* __restrict__ bias,
                                                float* __restrict__ out) {
  const int wave = blockIdx.x * 8 + (threadIdx.x >> 5);
  const int mt = wave / 16;          // 256 row tiles
  const int ns = wave % 16;          // 1024/64 col strips
  const int lane = threadIdx.x & 31;
  const int g = lane >> 4, c = lane & 15;

  const __bf16* abase = Oh + (size_t)mt * 16 * EMBED;
  const __bf16* bbase[4];
#pragma unroll
  for (int t = 0; t < 4; ++t) bbase[t] = wh + (size_t)(ns * 64 + t * 16) * EMBED;

  v8f acc[4];
#pragma unroll
  for (int t = 0; t < 4; ++t) zero8(acc[t]);

  v16bf a_cur = load_a_tile(abase, EMBED);
  v16bf b_cur[4];
#pragma unroll
  for (int t = 0; t < 4; ++t) b_cur[t] = load_b_tile(bbase[t], EMBED);

#pragma unroll 2
  for (int k = 32; k < EMBED; k += 32) {
    v16bf a_nxt = load_a_tile(abase + k, EMBED);
    v16bf b_nxt[4];
#pragma unroll
    for (int t = 0; t < 4; ++t) b_nxt[t] = load_b_tile(bbase[t] + k, EMBED);
#pragma unroll
    for (int t = 0; t < 4; ++t) acc[t] = wmma_bf16(a_cur, b_cur[t], acc[t]);
    a_cur = a_nxt;
#pragma unroll
    for (int t = 0; t < 4; ++t) b_cur[t] = b_nxt[t];
  }
#pragma unroll
  for (int t = 0; t < 4; ++t) acc[t] = wmma_bf16(a_cur, b_cur[t], acc[t]);

#pragma unroll
  for (int t = 0; t < 4; ++t) {
    const int n = ns * 64 + t * 16 + c;
    const float bv = bias[n];
#pragma unroll
    for (int r = 0; r < 8; ++r) {
      const int m = mt * 16 + r + 8 * g;
      out[(size_t)m * EMBED + n] = acc[t][r] + bv;
    }
  }
}

extern "C" void kernel_launch(void* const* d_in, const int* in_sizes, int n_in,
                              void* d_out, int out_size, void* d_ws, size_t ws_size,
                              hipStream_t stream) {
  const float* x     = (const float*)d_in[0];
  const int*   mask  = (const int*)d_in[1];
  const float* qkv_w = (const float*)d_in[2];
  const float* out_w = (const float*)d_in[3];
  const float* out_b = (const float*)d_in[4];
  float* out = (float*)d_out;

  char* ws = (char*)d_ws;
  size_t off = 0;
  auto alloc = [&](size_t elems) -> __bf16* {
    __bf16* p = (__bf16*)(ws + off);
    off += elems * sizeof(__bf16);
    return p;
  };

  const size_t nX  = (size_t)BATCH * SEQ * EMBED;        // 4,194,304
  const size_t nQW = (size_t)3 * EMBED * EMBED;          // 3,145,728
  const size_t nOW = (size_t)EMBED * EMBED;              // 1,048,576
  const size_t nH  = (size_t)BATCH * HEADS * SEQ * HDIM; // 4,194,304

  __bf16* xh  = alloc(nX);
  __bf16* qwh = alloc(nQW);
  __bf16* owh = alloc(nOW);
  __bf16* Qh  = alloc(nH);
  __bf16* Kh  = alloc(nH);
  __bf16* Vt  = alloc(nH);
  __bf16* Oh  = alloc(nX);

  cvt_f32_bf16<<<(int)(nX  / 256), 256, 0, stream>>>(x,     xh,  (int)nX);
  cvt_f32_bf16<<<(int)(nQW / 256), 256, 0, stream>>>(qkv_w, qwh, (int)nQW);
  cvt_f32_bf16<<<(int)(nOW / 256), 256, 0, stream>>>(out_w, owh, (int)nOW);

  // 256 row-tiles * 48 col-strips = 12288 waves / 8 waves-per-block
  qkv_gemm<<<1536, 256, 0, stream>>>(xh, qwh, Qh, Kh, Vt);

  attn_kernel<<<dim3(BATCH * HEADS, SEQ / 64), 128, 0, stream>>>(Qh, Kh, Vt, mask, Oh);

  // 256 row-tiles * 16 col-strips = 4096 waves / 8 per block
  out_gemm<<<512, 256, 0, stream>>>(Oh, owh, out_b, out);
}
